// LSTM_53549652247317
// MI455X (gfx1250) — matrix operand order repaired and checked by reading, not compiled, not run on hardware
//
#include <hip/hip_runtime.h>

// LSTM autoencoder: B=4096, T=128, F=64, H=32, G=4H=128
#define B_ 4096
#define T_ 128
#define F_ 64
#define H_ 32
#define G_ 128

#define WAVES 2   // 2 waves cooperate on one 16-row batch tile (split gate dim)
#define ROWS 16   // batch rows per block (WMMA M)

typedef __attribute__((ext_vector_type(16))) __bf16 v16bf;
typedef __attribute__((ext_vector_type(8)))  float  v8f;

union BFrag { v16bf v; unsigned u[8]; };

#if defined(__has_builtin) && __has_builtin(__builtin_amdgcn_cvt_pk_bf16_f32)
typedef __attribute__((ext_vector_type(2))) __bf16 v2bf;
__device__ __forceinline__ unsigned pack2_bf16(float lo, float hi) {
  v2bf r = __builtin_amdgcn_cvt_pk_bf16_f32(lo, hi);
  return __builtin_bit_cast(unsigned, r);
}
__device__ __forceinline__ unsigned short bf16_u16(float f) {
  return (unsigned short)pack2_bf16(f, 0.0f);
}
#else
// round-to-nearest (half-up) + single v_perm_b32 to gather both high halves
__device__ __forceinline__ unsigned pack2_bf16(float lo, float hi) {
  const unsigned ul = __float_as_uint(lo) + 0x8000u;
  const unsigned uh = __float_as_uint(hi) + 0x8000u;
  return __builtin_amdgcn_perm(uh, ul, 0x07060302u);  // {uh[31:16], ul[31:16]}
}
__device__ __forceinline__ unsigned short bf16_u16(float f) {
  return (unsigned short)((__float_as_uint(f) + 0x8000u) >> 16);
}
#endif

#if defined(__has_builtin) && __has_builtin(__builtin_amdgcn_tanhf)
__device__ __forceinline__ float sigmoidf_(float x) {
  return __builtin_fmaf(0.5f, __builtin_amdgcn_tanhf(0.5f * x), 0.5f);
}
#else
__device__ __forceinline__ float sigmoidf_(float x) {
  return __builtin_amdgcn_rcpf(1.0f + __expf(-x));
}
#endif
__device__ __forceinline__ float reluf_(float x) { return x > 0.0f ? x : 0.0f; }

__device__ __forceinline__ v8f wmma_bf16(const BFrag& a, const BFrag& b, v8f c) {
  return __builtin_amdgcn_wmma_f32_16x16x32_bf16(false, a.v, false, b.v,
                                                 (short)0, c, false, false);
}

__device__ __forceinline__ void load_bfrag(BFrag& f, const unsigned short* p) {
  const unsigned* q = (const unsigned*)p;   // 4-byte aligned (even k offset)
#pragma unroll
  for (int j = 0; j < 8; ++j) f.u[j] = q[j];
}

// A-fragment K is contiguous (k = 16*hi + s) -> use float4 (b128) loads
__device__ __forceinline__ void load_x_frags(BFrag xa[2], const float* xrow,
                                             int hi) {
#pragma unroll
  for (int ch = 0; ch < 2; ++ch)
#pragma unroll
    for (int jj = 0; jj < 4; ++jj) {
      const int k = 4 * jj + (hi ? 16 : 0);
      const float4 f4 = *(const float4*)(xrow + ch * 32 + k);
      xa[ch].u[2 * jj]     = pack2_bf16(f4.x, f4.y);
      xa[ch].u[2 * jj + 1] = pack2_bf16(f4.z, f4.w);
    }
}

__global__ __launch_bounds__(WAVES * 32)
void lstm_ae_kernel(const float* __restrict__ x,  const float* __restrict__ We,
                    const float* __restrict__ Ue, const float* __restrict__ be,
                    const float* __restrict__ Wd, const float* __restrict__ Ud,
                    const float* __restrict__ bd, const float* __restrict__ Wout,
                    const float* __restrict__ bout, float* __restrict__ out) {
  // Weights transposed to [N][K] bf16 so B-fragment K-pairs are contiguous u32s
  __shared__ unsigned short sWeT[G_ * F_];        // 16 KB
  __shared__ unsigned short sUeT[G_ * H_];        //  8 KB
  __shared__ unsigned short sUdT[G_ * H_];        //  8 KB
  __shared__ unsigned short sWdT[G_ * H_];        //  8 KB
  __shared__ unsigned short sWoT[F_ * H_];        //  4 KB
  __shared__ unsigned short sH[2][ROWS * H_];     // double-buffered h exchange

  const int tid = threadIdx.x;
  for (int i = tid; i < G_ * (F_ / 2); i += WAVES * 32) {
    int n = i / (F_ / 2), k = (i % (F_ / 2)) * 2;
    *(unsigned*)&sWeT[n * F_ + k] =
        pack2_bf16(We[k * G_ + n], We[(k + 1) * G_ + n]);
  }
  for (int i = tid; i < G_ * (H_ / 2); i += WAVES * 32) {
    int n = i / (H_ / 2), k = (i % (H_ / 2)) * 2;
    *(unsigned*)&sUeT[n * H_ + k] =
        pack2_bf16(Ue[k * G_ + n], Ue[(k + 1) * G_ + n]);
    *(unsigned*)&sUdT[n * H_ + k] =
        pack2_bf16(Ud[k * G_ + n], Ud[(k + 1) * G_ + n]);
    *(unsigned*)&sWdT[n * H_ + k] =
        pack2_bf16(Wd[k * G_ + n], Wd[(k + 1) * G_ + n]);
  }
  for (int i = tid; i < F_ * (H_ / 2); i += WAVES * 32) {
    int f = i / (H_ / 2), k = (i % (H_ / 2)) * 2;
    *(unsigned*)&sWoT[f * H_ + k] =
        pack2_bf16(Wout[k * F_ + f], Wout[(k + 1) * F_ + f]);
  }
  __syncthreads();

  const int lane = tid & 31;
  const int w    = tid >> 5;     // wave id: owns gate tiles {2q + w}
  const int m    = lane & 15;
  const int hi   = lane >> 4;
  const int b0   = blockIdx.x * ROWS;

  // A-matrix (16x32 bf16) K pair base per VGPR j (contiguous: k = 16*hi + 2*j)
  int kA[8];
#pragma unroll
  for (int j = 0; j < 8; ++j)
    kA[j] = 2 * j + (hi ? 16 : 0);
  const int kB = hi ? 16 : 0;  // B-matrix (32x16 bf16): k = kB + 2*j

  // bias splats for this wave's 4 gate tiles (q: 0=i,1=f,2=g,3=o)
  v8f beV[4], bdV[4];
#pragma unroll
  for (int q = 0; q < 4; ++q) {
    const float be_ = be[(2 * q + w) * 16 + m];
    const float bd_ = bd[(2 * q + w) * 16 + m];
#pragma unroll
    for (int e = 0; e < 8; ++e) { beV[q][e] = be_; bdV[q][e] = bd_; }
  }
  v8f boV[2];
#pragma unroll
  for (int q = 0; q < 2; ++q) {
    const float bo_ = bout[(2 * w + q) * 16 + m];
#pragma unroll
    for (int e = 0; e < 8; ++e) boV[q][e] = bo_;
  }

  // ================= encoder =================
  BFrag ueF[4];
#pragma unroll
  for (int q = 0; q < 4; ++q)
    load_bfrag(ueF[q], &sUeT[((2 * q + w) * 16 + m) * H_ + kB]);

  BFrag hA;
#pragma unroll
  for (int j = 0; j < 8; ++j) hA.u[j] = 0u;
  float cc[8];
#pragma unroll
  for (int v = 0; v < 8; ++v) cc[v] = 0.0f;

  const float* xbase = x + (size_t)(b0 + m) * T_ * F_;
  BFrag xa[2];
  load_x_frags(xa, xbase, hi);

#pragma unroll 2
  for (int t = 0; t < T_; ++t) {
    BFrag xn[2];
    if (t + 1 < T_) load_x_frags(xn, xbase + (size_t)(t + 1) * F_, hi);

    v8f acc[4];
#pragma unroll
    for (int q = 0; q < 4; ++q) {
      const int tl = 2 * q + w;
      BFrag wb;
      load_bfrag(wb, &sWeT[(tl * 16 + m) * F_ + kB]);
      v8f a = wmma_bf16(xa[0], wb, beV[q]);
      load_bfrag(wb, &sWeT[(tl * 16 + m) * F_ + 32 + kB]);
      a = wmma_bf16(xa[1], wb, a);
      acc[q] = wmma_bf16(hA, ueF[q], a);
    }
    // this wave updates h indices [16*w, 16*w+16); cell state stays local
    unsigned short* hbuf = sH[t & 1];
#pragma unroll
    for (int v = 0; v < 8; ++v) {
      const float i_ = sigmoidf_(acc[0][v]);
      const float f_ = sigmoidf_(acc[1][v]);
      const float g_ = reluf_(acc[2][v]);
      const float o_ = sigmoidf_(acc[3][v]);
      cc[v] = f_ * cc[v] + i_ * g_;
      const float h_ = o_ * reluf_(cc[v]);
      hbuf[(v + 8 * hi) * H_ + 16 * w + m] = bf16_u16(h_);
    }
    __syncthreads();   // exchange halves (double buffer -> one barrier/step)
#pragma unroll
    for (int j = 0; j < 8; ++j)
      hA.u[j] = *(const unsigned*)&hbuf[m * H_ + kA[j]];
    xa[0] = xn[0];
    xa[1] = xn[1];
  }

  // ================= decoder =================
  // RepeatVector: input projection hT@Wd+bd is timestep-invariant -> once
  v8f zc[4];
#pragma unroll
  for (int q = 0; q < 4; ++q) {
    BFrag wb;
    load_bfrag(wb, &sWdT[((2 * q + w) * 16 + m) * H_ + kB]);
    zc[q] = wmma_bf16(hA, wb, bdV[q]);
  }

  BFrag udF[4];
#pragma unroll
  for (int q = 0; q < 4; ++q)
    load_bfrag(udF[q], &sUdT[((2 * q + w) * 16 + m) * H_ + kB]);
  BFrag woF[2];
#pragma unroll
  for (int q = 0; q < 2; ++q)
    load_bfrag(woF[q], &sWoT[((2 * w + q) * 16 + m) * H_ + kB]);

#pragma unroll
  for (int j = 0; j < 8; ++j) hA.u[j] = 0u;
#pragma unroll
  for (int v = 0; v < 8; ++v) cc[v] = 0.0f;

#pragma unroll 2
  for (int t = 0; t < T_; ++t) {
    v8f acc[4];
#pragma unroll
    for (int q = 0; q < 4; ++q) acc[q] = wmma_bf16(hA, udF[q], zc[q]);

    unsigned short* hbuf = sH[t & 1];
#pragma unroll
    for (int v = 0; v < 8; ++v) {
      const float i_ = sigmoidf_(acc[0][v]);
      const float f_ = sigmoidf_(acc[1][v]);
      const float g_ = reluf_(acc[2][v]);
      const float o_ = sigmoidf_(acc[3][v]);
      cc[v] = f_ * cc[v] + i_ * g_;
      const float h_ = o_ * reluf_(cc[v]);
      hbuf[(v + 8 * hi) * H_ + 16 * w + m] = bf16_u16(h_);
    }
    __syncthreads();
#pragma unroll
    for (int j = 0; j < 8; ++j)
      hA.u[j] = *(const unsigned*)&hbuf[m * H_ + kA[j]];

    // out_t = h_t @ Wout + bout; this wave writes output cols [32w, 32w+32)
#pragma unroll
    for (int q = 0; q < 2; ++q) {
      const v8f oa = wmma_bf16(hA, woF[q], boV[q]);
#pragma unroll
      for (int v = 0; v < 8; ++v) {
        const int r = v + 8 * hi;
        out[((size_t)(b0 + r) * T_ + t) * F_ + (2 * w + q) * 16 + m] = oa[v];
      }
    }
  }
}

extern "C" void kernel_launch(void* const* d_in, const int* in_sizes, int n_in,
                              void* d_out, int out_size, void* d_ws, size_t ws_size,
                              hipStream_t stream) {
  (void)in_sizes; (void)n_in; (void)out_size; (void)d_ws; (void)ws_size;
  const float* x    = (const float*)d_in[0];
  const float* We   = (const float*)d_in[1];
  const float* Ue   = (const float*)d_in[2];
  const float* be   = (const float*)d_in[3];
  const float* Wd   = (const float*)d_in[4];
  const float* Ud   = (const float*)d_in[5];
  const float* bd   = (const float*)d_in[6];
  const float* Wout = (const float*)d_in[7];
  const float* bout = (const float*)d_in[8];
  float* out = (float*)d_out;

  dim3 grid(B_ / ROWS);      // 256 blocks, one 16-row tile each
  dim3 block(WAVES * 32);    // 2 waves split the gate dimension -> 512 waves
  lstm_ae_kernel<<<grid, block, 0, stream>>>(x, We, Ue, be, Wd, Ud, bd, Wout,
                                             bout, out);
}
